// Mamba_75505525063753
// MI455X (gfx1250) — matrix operand order repaired and checked by reading, not compiled, not run on hardware
//
#include <hip/hip_runtime.h>
#include <hip/hip_bf16.h>
#include <math.h>

// Problem constants (from reference)
#define BBATCH 4
#define LLEN   2048
#define DMODEL 256
#define DINNER 512
#define NSTATE 16
#define RRANK  16
#define KCONV  4
#define NLAYER 2
#define EPSF   1e-5f

typedef __attribute__((ext_vector_type(16))) _Float16 v16h;
typedef __attribute__((ext_vector_type(8)))  _Float16 v8h;
typedef __attribute__((ext_vector_type(4)))  _Float16 v4h;
typedef __attribute__((ext_vector_type(8)))  float    v8f;

// ---------------------------------------------------------------------------
// Block-wide reduction (256 threads = 8 wave32)
// ---------------------------------------------------------------------------
__device__ __forceinline__ float block_reduce(float v, int is_max) {
    __shared__ float sh[8];
    __shared__ float res;
    int lane = threadIdx.x & 31;
    int w    = threadIdx.x >> 5;
    #pragma unroll
    for (int o = 16; o > 0; o >>= 1) {
        float t = __shfl_down(v, o, 32);
        v = is_max ? fmaxf(v, t) : (v + t);
    }
    __syncthreads();                 // protect sh across repeated calls
    if (lane == 0) sh[w] = v;
    __syncthreads();
    if (threadIdx.x == 0) {
        float r = sh[0];
        #pragma unroll
        for (int i = 1; i < 8; ++i) r = is_max ? fmaxf(r, sh[i]) : (r + sh[i]);
        res = r;
    }
    __syncthreads();
    return res;
}

// ---------------------------------------------------------------------------
// RMSNorm: one block (256 threads) per row of DM=256
// ---------------------------------------------------------------------------
__global__ __launch_bounds__(256) void rmsnorm_kernel(
    const float* __restrict__ x, const float* __restrict__ w,
    float* __restrict__ o)
{
    long row = blockIdx.x;
    float v = x[row * DMODEL + threadIdx.x];
    float s = block_reduce(v * v, 0);
    float inv = rsqrtf(s * (1.0f / DMODEL) + EPSF);
    o[row * DMODEL + threadIdx.x] = v * inv * w[threadIdx.x];
}

// ---------------------------------------------------------------------------
// Generic batched GEMM via V_WMMA_F32_16X16X32_F16, templated on layout:
//   TRANS = 1 : Bmat[k][n] = W[n*ldw + k]   (y = x @ W^T, W stored (N,K))
//   TRANS = 0 : Bmat[k][n] = W[k*ldw + n]   (y = x @ W,   W stored (K,N))
//   GUARD = 1 : N not a multiple of the 128-wide block tile (only N=48 case)
// Block tile 64(M) x 128(N), K-step 32, 8 waves (2x4), 2x2 16x16 WMMA/wave.
// Register double-buffer pipeline: next tile's global loads are issued before
// the current tile's WMMAs so VMEM latency overlaps the matrix pipe.
// Requirements at every call site: M % 64 == 0, Kd % 32 == 0, lda/ldw % 4 == 0.
// ---------------------------------------------------------------------------
#define GBM 64
#define GBN 128
#define GBK 32
#define SAS 48   // padded LDS row stride (halves): 96B, conflict-reducing
#define SBS 48

__device__ __forceinline__ v4h cvt4(float a, float b, float c, float d) {
    v4h h;
    h[0] = (_Float16)a; h[1] = (_Float16)b;
    h[2] = (_Float16)c; h[3] = (_Float16)d;
    return h;
}

template <int TRANS, int GUARD>
__global__ __launch_bounds__(256) void gemm_wmma_t(
    const float* __restrict__ A, long lda, long sA,
    const float* __restrict__ W, long ldw, long sW,
    const float* __restrict__ bias,
    const float* __restrict__ resid,
    float* __restrict__ C, long ldc, long sC,
    int N, int Kd, int act)
{
    __shared__ __align__(32) _Float16 sa[GBM * SAS];
    __shared__ __align__(32) _Float16 sb[GBN * SBS];

    const int tid  = threadIdx.x;
    const int wave = tid >> 5;
    const int lane = tid & 31;
    const int wm   = wave >> 2;   // 0..1 -> 32 rows each
    const int wn   = wave & 3;    // 0..3 -> 32 cols each
    const int lhalf = lane & 15;
    const int lgrp  = lane >> 4;

    const long bz = blockIdx.z;
    const float* Ab = A + bz * sA;
    const float* Wb = W + bz * sW;
    float*       Cb = C + bz * sC;
    const int m0 = blockIdx.y * GBM;
    const int n0 = blockIdx.x * GBN;

    // per-thread staging coordinates
    const int nrow  = tid & 127;          // !TRANS: fixed n column
    const int kbase = (tid >> 7) * 16;    // !TRANS: 16 k's per thread

    float4 fa[2];
    float4 fb4[4];
    float  fbn[16];

    auto load_tiles = [&](int k0) {
        #pragma unroll
        for (int i = 0; i < 2; ++i) {
            int idx4 = tid + i * 256;              // 64 rows x 8 float4
            int r = idx4 >> 3, c = (idx4 & 7) * 4;
            fa[i] = *(const float4*)(Ab + (long)(m0 + r) * lda + k0 + c);
        }
        if (TRANS) {
            #pragma unroll
            for (int i = 0; i < 4; ++i) {
                int idx4 = tid + i * 256;          // 128 rows x 8 float4
                int r = idx4 >> 3, c = (idx4 & 7) * 4;
                if (!GUARD || (n0 + r) < N)
                    fb4[i] = *(const float4*)(Wb + (long)(n0 + r) * ldw + k0 + c);
                else
                    fb4[i] = make_float4(0.f, 0.f, 0.f, 0.f);
            }
        } else {
            const float* wp = Wb + (long)(k0 + kbase) * ldw + n0 + nrow;
            #pragma unroll
            for (int j = 0; j < 16; ++j) fbn[j] = wp[(long)j * ldw];
        }
    };
    auto store_tiles = [&]() {
        #pragma unroll
        for (int i = 0; i < 2; ++i) {
            int idx4 = tid + i * 256;
            int r = idx4 >> 3, c = (idx4 & 7) * 4;
            *(v4h*)&sa[r * SAS + c] = cvt4(fa[i].x, fa[i].y, fa[i].z, fa[i].w);
        }
        if (TRANS) {
            #pragma unroll
            for (int i = 0; i < 4; ++i) {
                int idx4 = tid + i * 256;
                int r = idx4 >> 3, c = (idx4 & 7) * 4;
                *(v4h*)&sb[r * SBS + c] = cvt4(fb4[i].x, fb4[i].y, fb4[i].z, fb4[i].w);
            }
        } else {
            #pragma unroll
            for (int j = 0; j < 4; ++j)
                *(v4h*)&sb[nrow * SBS + kbase + j * 4] =
                    cvt4(fbn[j * 4 + 0], fbn[j * 4 + 1], fbn[j * 4 + 2], fbn[j * 4 + 3]);
        }
    };

    v8f acc[2][2] = {};
    load_tiles(0);

    for (int k0 = 0; k0 < Kd; k0 += GBK) {
        store_tiles();
        __syncthreads();
        if (k0 + GBK < Kd) load_tiles(k0 + GBK);   // overlap with WMMAs below

        // ---- fragments per ISA 16-bit layouts ----
        // A 16x32: lanes0-15 M=lane, K 0-7 & 16-23; lanes16-31 M=lane-16, K 8-15 & 24-31
        v16h af[2];
        #pragma unroll
        for (int mi = 0; mi < 2; ++mi) {
            int row = wm * 32 + mi * 16 + lhalf;
            const _Float16* p = &sa[row * SAS + lgrp * 8];
            v8h lo = *(const v8h*)p;
            v8h hi = *(const v8h*)(p + 16);
            #pragma unroll
            for (int q = 0; q < 8; ++q) { af[mi][q] = lo[q]; af[mi][q + 8] = hi[q]; }
        }
        // B 32x16: lanes0-15 N=lane K 0-7 & 16-23; lanes16-31 N=lane-16 K 8-15 & 24-31
        v16h bf[2];
        #pragma unroll
        for (int ni = 0; ni < 2; ++ni) {
            int rowb = wn * 32 + ni * 16 + lhalf;
            const _Float16* pb = &sb[rowb * SBS + lgrp * 8];
            v8h blo = *(const v8h*)pb;
            v8h bhi = *(const v8h*)(pb + 16);
            #pragma unroll
            for (int q = 0; q < 8; ++q) { bf[ni][q] = blo[q]; bf[ni][q + 8] = bhi[q]; }
        }

        #pragma unroll
        for (int mi = 0; mi < 2; ++mi)
            #pragma unroll
            for (int ni = 0; ni < 2; ++ni)
                acc[mi][ni] = __builtin_amdgcn_wmma_f32_16x16x32_f16(
                    false, af[mi], false, bf[ni], (short)0, acc[mi][ni],
                    false, false);
        __syncthreads();
    }

    // ---- epilogue: C 16x16 f32 layout (VGPR r: lanes0-15 M=r, lanes16-31 M=8+r)
    #pragma unroll
    for (int mi = 0; mi < 2; ++mi) {
        #pragma unroll
        for (int ni = 0; ni < 2; ++ni) {
            int gn = n0 + wn * 32 + ni * 16 + lhalf;
            if (GUARD && gn >= N) continue;
            float bv = bias ? bias[gn] : 0.0f;
            #pragma unroll
            for (int r = 0; r < 8; ++r) {
                int gm = m0 + wm * 32 + mi * 16 + r + lgrp * 8;
                float v = acc[mi][ni][r] + bv;
                if (act == 1)       v = v / (1.0f + __expf(-v));             // silu
                else if (act == 2)  v = (v > 20.0f) ? v : log1pf(__expf(v)); // softplus
                if (resid) v += resid[bz * sC + (long)gm * ldc + gn];
                Cb[(long)gm * ldc + gn] = v;
            }
        }
    }
}

// host-side dispatch over the template variants
static inline void launch_gemm(dim3 grid, hipStream_t stream,
                               const float* A, long lda, long sA,
                               const float* W, long ldw, long sW, int trans,
                               const float* bias, const float* resid,
                               float* C, long ldc, long sC,
                               int N, int Kd, int act)
{
    dim3 blk(256);
    if (trans) {
        if (N % GBN == 0)
            gemm_wmma_t<1, 0><<<grid, blk, 0, stream>>>(A, lda, sA, W, ldw, sW,
                bias, resid, C, ldc, sC, N, Kd, act);
        else
            gemm_wmma_t<1, 1><<<grid, blk, 0, stream>>>(A, lda, sA, W, ldw, sW,
                bias, resid, C, ldc, sC, N, Kd, act);
    } else {
        gemm_wmma_t<0, 0><<<grid, blk, 0, stream>>>(A, lda, sA, W, ldw, sW,
            bias, resid, C, ldc, sC, N, Kd, act);
    }
}

// ---------------------------------------------------------------------------
// Depthwise causal conv (K=4) + SiLU. rev=1 reads the time-flipped input.
// ---------------------------------------------------------------------------
__global__ __launch_bounds__(256) void conv_silu_kernel(
    const float* __restrict__ xp, const float* __restrict__ w,
    const float* __restrict__ bias, float* __restrict__ o, int rev)
{
    long i = (long)blockIdx.x * 256 + threadIdx.x;
    int d = (int)(i & (DINNER - 1));
    long pos = i >> 9;                       // DI = 512
    int l = (int)(pos & (LLEN - 1));
    int b = (int)(pos >> 11);                // L = 2048
    float acc = bias[d];
    #pragma unroll
    for (int j = 0; j < KCONV; ++j) {
        int src = l - (KCONV - 1) + j;
        if (src >= 0) {
            int sl = rev ? (LLEN - 1 - src) : src;
            acc += xp[((long)b * LLEN + sl) * DINNER + d] * w[d * KCONV + j];
        }
    }
    o[((long)b * LLEN + l) * DINNER + d] = acc / (1.0f + __expf(-acc));
}

// ---------------------------------------------------------------------------
// Fused selective scan: per (b, d) thread keeps h[16] in registers, walks L.
// delta = softplus(dlt @ w_dt^T + b_dt) computed in-register (R=16).
// grid: (DI/256, B), block 256.
// ---------------------------------------------------------------------------
__global__ __launch_bounds__(256) void ssm_scan_kernel(
    const float* __restrict__ xdir, const float* __restrict__ dbc,
    const float* __restrict__ w_dt, const float* __restrict__ b_dt,
    const float* __restrict__ A_log, const float* __restrict__ Dp,
    float* __restrict__ ydir, int ostride)
{
    const int tid = threadIdx.x;
    const int d = blockIdx.x * 256 + tid;
    const int b = blockIdx.y;
    __shared__ float sh[RRANK + 2 * NSTATE];   // dlt | B | C of current step

    float a[NSTATE], h[NSTATE], wdt[RRANK];
    #pragma unroll
    for (int n = 0; n < NSTATE; ++n) {
        a[n] = -__expf(A_log[(long)d * NSTATE + n]);
        h[n] = 0.0f;
    }
    #pragma unroll
    for (int r = 0; r < RRANK; ++r) wdt[r] = w_dt[(long)d * RRANK + r];
    const float bdt = b_dt[d];
    const float dp  = Dp[d];

    for (int l = 0; l < LLEN; ++l) {
        long row = (long)b * LLEN + l;
        if (tid < RRANK + 2 * NSTATE) sh[tid] = dbc[row * 48 + tid];
        __syncthreads();
        float u = bdt;
        #pragma unroll
        for (int r = 0; r < RRANK; ++r) u += sh[r] * wdt[r];
        float dt = (u > 20.0f) ? u : log1pf(__expf(u));
        float xt = xdir[row * DINNER + d];
        float dx = dt * xt;
        float y = 0.0f;
        #pragma unroll
        for (int n = 0; n < NSTATE; ++n) {
            float dA = __expf(dt * a[n]);
            h[n] = dA * h[n] + dx * sh[RRANK + n];
            y += h[n] * sh[RRANK + NSTATE + n];
        }
        ydir[row * (long)ostride + d] = y + dp * xt;
        __syncthreads();
    }
}

// ---------------------------------------------------------------------------
// Row softmax (rows of length n, in place), block 256
// ---------------------------------------------------------------------------
__global__ __launch_bounds__(256) void softmax_kernel(float* __restrict__ x, int n)
{
    long row = blockIdx.x;
    float* p = x + row * (long)n;
    float mx = -3.4e38f;
    for (int i = threadIdx.x; i < n; i += 256) mx = fmaxf(mx, p[i]);
    mx = block_reduce(mx, 1);
    float s = 0.0f;
    for (int i = threadIdx.x; i < n; i += 256) {
        float e = __expf(p[i] - mx);
        p[i] = e;
        s += e;
    }
    s = block_reduce(s, 0);
    float inv = 1.0f / s;
    for (int i = threadIdx.x; i < n; i += 256) p[i] *= inv;
}

// ---------------------------------------------------------------------------
// Gate: g = ytok * silu(z)
// ---------------------------------------------------------------------------
__global__ __launch_bounds__(256) void gate_kernel(
    const float* __restrict__ ytok, const float* __restrict__ z,
    float* __restrict__ g)
{
    long i = (long)blockIdx.x * 256 + threadIdx.x;
    float zv = z[i];
    g[i] = ytok[i] * (zv / (1.0f + __expf(-zv)));
}

// ---------------------------------------------------------------------------
// Host launcher
// ---------------------------------------------------------------------------
extern "C" void kernel_launch(void* const* d_in, const int* in_sizes, int n_in,
                              void* d_out, int out_size, void* d_ws, size_t ws_size,
                              hipStream_t stream)
{
    (void)in_sizes; (void)n_in; (void)out_size; (void)ws_size;
    const float* x        = (const float*)d_in[0];
    const float* norm_w   = (const float*)d_in[1];
    const float* w_in_x   = (const float*)d_in[2];
    const float* b_in_x   = (const float*)d_in[3];
    const float* w_in_z   = (const float*)d_in[4];
    const float* b_in_z   = (const float*)d_in[5];
    const float* conv_wf  = (const float*)d_in[6];
    const float* conv_bf  = (const float*)d_in[7];
    const float* conv_wb  = (const float*)d_in[8];
    const float* conv_bb  = (const float*)d_in[9];
    const float* w_xproj  = (const float*)d_in[10];
    const float* b_xproj  = (const float*)d_in[11];
    const float* w_dt     = (const float*)d_in[12];
    const float* b_dt     = (const float*)d_in[13];
    const float* A_log    = (const float*)d_in[14];
    const float* Dp       = (const float*)d_in[15];
    const float* token_wA = (const float*)d_in[16];
    const float* token_wV = (const float*)d_in[17];
    const float* w_pro    = (const float*)d_in[18];
    const float* b_pro    = (const float*)d_in[19];
    const float* w_out    = (const float*)d_in[20];
    const float* b_out    = (const float*)d_in[21];
    float* out = (float*)d_out;

    const size_t MROWS = (size_t)BBATCH * LLEN;   // 8192
    size_t off = 0;
    auto alloc = [&](size_t nf) {
        float* p = (float*)((char*)d_ws + off);
        off += ((nf * sizeof(float)) + 255) & ~(size_t)255;
        return p;
    };
    float* nx    = alloc(MROWS * DMODEL);       // reused as dbc later
    float* xp    = alloc(MROWS * DINNER);
    float* zb    = alloc(MROWS * DINNER);
    float* xdir  = alloc(MROWS * DINNER);
    float* y2    = alloc(MROWS * 2 * DINNER);   // [yf | yb] concatenated
    float* ypro  = alloc(MROWS * DINNER);       // reused as gate output
    float* yv    = alloc(MROWS * DINNER);
    float* logit = alloc((size_t)BBATCH * LLEN * LLEN);
    float* ytok  = alloc(MROWS * DINNER);
    float* dbc   = nx;

    hipMemcpyAsync(out, x, MROWS * DMODEL * sizeof(float),
                   hipMemcpyDeviceToDevice, stream);

    dim3 blk(256);
    for (int l = 0; l < NLAYER; ++l) {
        // nx = rmsnorm(x) * norm_w[l]
        rmsnorm_kernel<<<(unsigned)MROWS, blk, 0, stream>>>(out, norm_w + l * DMODEL, nx);

        // xp = nx @ w_in_x^T + b ; z = nx @ w_in_z^T + b
        launch_gemm(dim3(DINNER / GBN, MROWS / GBM, 1), stream,
            nx, DMODEL, 0, w_in_x + (size_t)l * DINNER * DMODEL, DMODEL, 0, 1,
            b_in_x + l * DINNER, nullptr, xp, DINNER, 0, DINNER, DMODEL, 0);
        launch_gemm(dim3(DINNER / GBN, MROWS / GBM, 1), stream,
            nx, DMODEL, 0, w_in_z + (size_t)l * DINNER * DMODEL, DMODEL, 0, 1,
            b_in_z + l * DINNER, nullptr, zb, DINNER, 0, DINNER, DMODEL, 0);

        // forward + backward SSM branches
        for (int dir = 0; dir < 2; ++dir) {
            const float* cw = dir ? conv_wb + (size_t)l * DINNER * KCONV
                                  : conv_wf + (size_t)l * DINNER * KCONV;
            const float* cb = dir ? conv_bb + l * DINNER : conv_bf + l * DINNER;
            conv_silu_kernel<<<(unsigned)(MROWS * DINNER / 256), blk, 0, stream>>>(
                xp, cw, cb, xdir, dir);
            // dbc = xdir @ w_xproj^T + b_xproj   (N = 48 -> guarded variant)
            launch_gemm(dim3(1, MROWS / GBM, 1), stream,
                xdir, DINNER, 0, w_xproj + (size_t)l * 48 * DINNER, DINNER, 0, 1,
                b_xproj + l * 48, nullptr, dbc, 48, 0, 48, DINNER, 0);
            ssm_scan_kernel<<<dim3(DINNER / 256, BBATCH), blk, 0, stream>>>(
                xdir, dbc, w_dt + (size_t)l * DINNER * RRANK, b_dt + l * DINNER,
                A_log + (size_t)l * DINNER * NSTATE, Dp + l * DINNER,
                y2 + dir * DINNER, 2 * DINNER);
        }

        // ypro = [yf|yb] @ w_pro^T + b_pro
        launch_gemm(dim3(DINNER / GBN, MROWS / GBM, 1), stream,
            y2, 2 * DINNER, 0, w_pro + (size_t)l * DINNER * 2 * DINNER,
            2 * DINNER, 0, 1,
            b_pro + l * DINNER, nullptr, ypro, DINNER, 0, DINNER, 2 * DINNER, 0);

        // yv = ypro @ wV   (wV stored (K,N))
        launch_gemm(dim3(DINNER / GBN, MROWS / GBM, 1), stream,
            ypro, DINNER, 0, token_wV + (size_t)l * DINNER * DINNER, DINNER, 0, 0,
            nullptr, nullptr, yv, DINNER, 0, DINNER, DINNER, 0);

        // logits[b,t,l] = wA[t,:] . ypro[b,l,:]   (batched, B stored (N,K))
        launch_gemm(dim3(LLEN / GBN, LLEN / GBM, BBATCH), stream,
            token_wA + (size_t)l * LLEN * DINNER, DINNER, 0,
            ypro, DINNER, (long)LLEN * DINNER, 1,
            nullptr, nullptr, logit, LLEN, (long)LLEN * LLEN, LLEN, DINNER, 0);

        softmax_kernel<<<(unsigned)(BBATCH * LLEN), blk, 0, stream>>>(logit, LLEN);

        // ytok[b,t,d] = Aw[b,t,:] @ yv[b,:,d]
        launch_gemm(dim3(DINNER / GBN, LLEN / GBM, BBATCH), stream,
            logit, LLEN, (long)LLEN * LLEN,
            yv, DINNER, (long)LLEN * DINNER, 0,
            nullptr, nullptr, ytok, DINNER, (long)LLEN * DINNER, DINNER, LLEN, 0);

        // gate: ypro <- ytok * silu(z)
        gate_kernel<<<(unsigned)(MROWS * DINNER / 256), blk, 0, stream>>>(ytok, zb, ypro);

        // x <- gate @ w_out^T + b_out + x   (residual, in place on out)
        launch_gemm(dim3(DMODEL / GBN, MROWS / GBM, 1), stream,
            ypro, DINNER, 0, w_out + (size_t)l * DMODEL * DINNER, DINNER, 0, 1,
            b_out + l * DMODEL, out, out, DMODEL, 0, DMODEL, DINNER, 0);
    }
}
